// MultiHeadAttentionLayer_18056042512404
// MI455X (gfx1250) — matrix operand order repaired and checked by reading, not compiled, not run on hardware
//
#include <hip/hip_runtime.h>
#include <math.h>

// MHA forward for MI455X (gfx1250, wave32, WMMA).
//   wcvt_kernel x3 : W(1024x1024 f32) -> Wt[n][k] bf16 (LDS-tiled transpose)
//   proj_kernel x3 : Q = q@Wq+bq, K = k@Wq+bq (faithful to ref), V = v@Wv+bv
//                    fp32 in -> bf16 workspace; Q/K head-split (B,H,S,64),
//                    V stored transposed (B,H,64,S).
//   attn_kernel    : flash attention, bf16 WMMA, f32 accum; K/V key tiles
//                    double-buffered in LDS via GLOBAL_LOAD_ASYNC_TO_LDS_B128
//                    (ASYNCcnt) when available, sync LDS copy otherwise.
//   oproj_kernel   : out = merge_heads(O) @ Wo + bo -> fp32 d_out
// Workspace: 4 * B*H*S*64 bf16 (32 MB) + 3 * D*D bf16 (6 MB) = 38 MB.

#define B_ 2
#define S_ 2048
#define D_ 1024
#define H_ 16
#define DEPTH_ 64

typedef __attribute__((ext_vector_type(16))) __bf16 v16bf;
typedef __attribute__((ext_vector_type(8)))  float  v8f;
typedef int v4i_vs __attribute__((vector_size(16)));   // matches builtin proto

union FragBF {
  v16bf v;
  __bf16 e[16];
  unsigned short u[16];
  uint4 q[2];          // q[0] = elements 0..7, q[1] = 8..15
};

// A-operand k index for element j (16-bit A 16x32 layout, ISA 7.12.2):
//   j in 0..7  -> k = 8*half + j          (contiguous run of 8)
//   j in 8..15 -> k = 16 + 8*half + (j-8) (contiguous run of 8)
// B-operand k index: same as A (runs of 8 at k = 8*half and 16 + 8*half)

#if defined(__has_builtin)
#if __has_builtin(__builtin_amdgcn_global_load_async_to_lds_b128) && \
    __has_builtin(__builtin_amdgcn_s_wait_asynccnt)
#define USE_ASYNC_LDS 1
#endif
#endif

// Copy 16 bytes global -> LDS. Async (ASYNCcnt-tracked) when the toolchain
// exposes the gfx1250 builtin; synchronous VGPR round-trip otherwise.
__device__ __forceinline__ void copy_b128_to_lds(const void* g, void* l) {
#ifdef USE_ASYNC_LDS
  __builtin_amdgcn_global_load_async_to_lds_b128(
      (__attribute__((address_space(1))) v4i_vs*)(uintptr_t)g,
      (__attribute__((address_space(3))) v4i_vs*)(unsigned)(uintptr_t)l,
      0, 0);
#else
  *(uint4*)l = *(const uint4*)g;
#endif
}
__device__ __forceinline__ void async_copies_wait() {
#ifdef USE_ASYNC_LDS
  __builtin_amdgcn_s_wait_asynccnt(0);
#endif
}

// ---------------------------------------------------------------------------
// Weight transpose + bf16 convert: Wt[n][k] = (bf16)W[k][n], 32x32 LDS tiles.
// ---------------------------------------------------------------------------
__global__ __launch_bounds__(256) void wcvt_kernel(
    const float* __restrict__ W, unsigned short* __restrict__ Wt) {
  __shared__ __align__(16) unsigned short tile[32][33];
  const int tx = threadIdx.x & 31;
  const int ty = threadIdx.x >> 5;          // 0..7
  const int k0 = blockIdx.x * 32;
  const int n0 = blockIdx.y * 32;
#pragma unroll
  for (int i = 0; i < 32; i += 8) {
    FragBF t;
    t.e[0] = (__bf16)W[(size_t)(k0 + ty + i) * D_ + n0 + tx];
    tile[ty + i][tx] = t.u[0];
  }
  __syncthreads();
#pragma unroll
  for (int i = 0; i < 32; i += 8)
    Wt[(size_t)(n0 + ty + i) * D_ + k0 + tx] = tile[tx][ty + i];
}

// ---------------------------------------------------------------------------
// Projection GEMM: Y(4096x1024) = X @ W + bias -> bf16 with caller strides.
// B operand from pre-transposed bf16 Wt[n][k]: contiguous b128 loads.
// Block = 256 threads = 8 waves, each wave one 16x16 tile (2x4 wave grid).
// ---------------------------------------------------------------------------
__global__ __launch_bounds__(256) void proj_kernel(
    const float* __restrict__ X, const unsigned short* __restrict__ Wt,
    const float* __restrict__ bias, unsigned short* __restrict__ Yh,
    int sStride, int dStride) {
  const int lane  = threadIdx.x & 31;
  const int wave  = threadIdx.x >> 5;
  const int half  = lane >> 4;
  const int col16 = lane & 15;
  const int rowBase = blockIdx.x * 32 + (wave >> 2) * 16;
  const int colBase = blockIdx.y * 64 + (wave & 3) * 16;
  const int m = rowBase + col16;   // A row this lane loads
  const int n = colBase + col16;   // B col this lane loads / C col

  v8f acc = {};
  for (int kb = 0; kb < D_; kb += 32) {
    FragBF a, b;
    // A: two contiguous 8-float runs -> 4x float4 loads + packed cvt
    const float4* xr = (const float4*)(X + (size_t)m * D_ + kb + 8 * half);
    float4 x0 = xr[0], x1 = xr[1];       // k = kb+8*half .. +7
    float4 x2 = xr[4], x3 = xr[5];       // k = kb+16+8*half .. +7
    a.e[0] = (__bf16)x0.x; a.e[1] = (__bf16)x0.y; a.e[2] = (__bf16)x0.z; a.e[3] = (__bf16)x0.w;
    a.e[4] = (__bf16)x1.x; a.e[5] = (__bf16)x1.y; a.e[6] = (__bf16)x1.z; a.e[7] = (__bf16)x1.w;
    a.e[8] = (__bf16)x2.x; a.e[9] = (__bf16)x2.y; a.e[10] = (__bf16)x2.z; a.e[11] = (__bf16)x2.w;
    a.e[12] = (__bf16)x3.x; a.e[13] = (__bf16)x3.y; a.e[14] = (__bf16)x3.z; a.e[15] = (__bf16)x3.w;
    // B: Wt[n][kb + 8*half + j] and Wt[n][kb + 16 + 8*half + j] -> two 8-elem runs
    const uint4* wt = (const uint4*)(Wt + (size_t)n * D_ + kb + 8 * half);
    b.q[0] = wt[0]; b.q[1] = wt[2];

    acc = __builtin_amdgcn_wmma_f32_16x16x32_bf16(
        false, a.v, false, b.v, (short)0, acc, false, false);
  }
  const float bn = bias[n];
  const int h = n >> 6, d = n & 63;
  unsigned short* yb = Yh + (size_t)d * dStride;
#pragma unroll
  for (int r = 0; r < 8; ++r) {
    int row = rowBase + r + (half << 3);           // [0,4096)
    int bb = row >> 11, s = row & (S_ - 1);
    FragBF t; t.e[0] = (__bf16)(acc[r] + bn);
    yb[(((size_t)bb * H_ + h) * S_ * DEPTH_) + (size_t)s * sStride] = t.u[0];
  }
}

// ---------------------------------------------------------------------------
// Flash attention: one wave = 16 query rows, key tiles of 32 keys.
// Transposed logits S^T = K(A) x Q^T(B): the softmaxed C-fragment maps
// element-for-element onto the A-fragment of P for P@V (no lane transpose).
// K (32x64) and V (64x32, d-major) tiles are double-buffered in LDS,
// staged with async b128 global->LDS transfers (one per thread per tile).
// ---------------------------------------------------------------------------
__global__ __launch_bounds__(256) void attn_kernel(
    const unsigned short* __restrict__ Qb, const unsigned short* __restrict__ Kb,
    const unsigned short* __restrict__ Vt, const float* __restrict__ mask,
    unsigned short* __restrict__ Ob) {
  __shared__ __align__(16) unsigned short KtL[2][32][64];   // [buf][key][d]
  __shared__ __align__(16) unsigned short VtL[2][64][32];   // [buf][d][key]

  const int tid   = threadIdx.x;
  const int lane  = tid & 31;
  const int wave  = tid >> 5;
  const int half  = lane >> 4;
  const int col16 = lane & 15;
  const int bh    = blockIdx.y;               // 0..31
  const int bidx  = bh >> 4;                  // batch
  const int qBase = blockIdx.x * 128 + wave * 16;

  const unsigned short* Qh = Qb + (size_t)bh * S_ * DEPTH_;
  const unsigned short* Kh = Kb + (size_t)bh * S_ * DEPTH_;
  const unsigned short* Vh = Vt + (size_t)bh * S_ * DEPTH_;  // (64, 2048) d-major
  const float* mb = mask + (size_t)bidx * S_;

  // staging assignment: one b128 per thread per tile for K and for V
  const int krow = tid >> 3, kseg = tid & 7;   // K: 32 rows x 8 segs of 8 bf16
  const int vrow = tid >> 2, vseg = tid & 3;   // V: 64 rows x 4 segs of 8 bf16

  // Q^T as B operand: runs of 8 at k = 8*half and 16 + 8*half
  FragBF bq0, bq1;
  const int mq = qBase + col16;
  {
    const uint4* qp = (const uint4*)(Qh + (size_t)mq * DEPTH_ + 8 * half);
    bq0.q[0] = qp[0]; bq0.q[1] = qp[2];        // d = 8*half .. +7, 16+8*half .. +7
    const uint4* qp1 = (const uint4*)(Qh + (size_t)mq * DEPTH_ + 32 + 8 * half);
    bq1.q[0] = qp1[0]; bq1.q[1] = qp1[2];      // d = 32+8*half .. +7, 48+8*half .. +7
  }

  // prologue: stage tile 0 into buffer 0
  copy_b128_to_lds(Kh + (size_t)krow * DEPTH_ + kseg * 8, &KtL[0][krow][kseg * 8]);
  copy_b128_to_lds(Vh + (size_t)vrow * S_ + vseg * 8, &VtL[0][vrow][vseg * 8]);
  async_copies_wait();
  __syncthreads();

  const float scale = 0.125f;   // 1/sqrt(64)
  float m_run = -INFINITY, l_run = 0.f;
  v8f o0 = {}, o1 = {}, o2 = {}, o3 = {};
  const int NT = S_ / 32;

  for (int kt = 0; kt < NT; ++kt) {
    const int key0 = kt * 32;
    const int buf  = kt & 1;

    // stage next tile into the other buffer (overlaps with compute below);
    // barrier at end of previous iteration guarantees its readers are done
    if (kt + 1 < NT) {
      const int nk0 = key0 + 32;
      copy_b128_to_lds(Kh + (size_t)(nk0 + krow) * DEPTH_ + kseg * 8,
                       &KtL[buf ^ 1][krow][kseg * 8]);
      copy_b128_to_lds(Vh + (size_t)vrow * S_ + nk0 + vseg * 8,
                       &VtL[buf ^ 1][vrow][vseg * 8]);
    }

    // K rows as A operand from LDS: runs of 8 at d = 8*half(+16)(+32)(+48)
    FragBF a00, a01, a10, a11;
    a00.q[0] = *(const uint4*)&KtL[buf][col16][8 * half];
    a00.q[1] = *(const uint4*)&KtL[buf][col16][16 + 8 * half];
    a01.q[0] = *(const uint4*)&KtL[buf][col16][32 + 8 * half];
    a01.q[1] = *(const uint4*)&KtL[buf][col16][48 + 8 * half];
    a10.q[0] = *(const uint4*)&KtL[buf][col16 + 16][8 * half];
    a10.q[1] = *(const uint4*)&KtL[buf][col16 + 16][16 + 8 * half];
    a11.q[0] = *(const uint4*)&KtL[buf][col16 + 16][32 + 8 * half];
    a11.q[1] = *(const uint4*)&KtL[buf][col16 + 16][48 + 8 * half];

    v8f t0 = {}, t1 = {};
    t0 = __builtin_amdgcn_wmma_f32_16x16x32_bf16(false, a00.v, false, bq0.v, (short)0, t0, false, false);
    t0 = __builtin_amdgcn_wmma_f32_16x16x32_bf16(false, a01.v, false, bq1.v, (short)0, t0, false, false);
    t1 = __builtin_amdgcn_wmma_f32_16x16x32_bf16(false, a10.v, false, bq0.v, (short)0, t1, false, false);
    t1 = __builtin_amdgcn_wmma_f32_16x16x32_bf16(false, a11.v, false, bq1.v, (short)0, t1, false, false);

    // scale + mask; tile max over this lane's 16 keys, then pair lanes
    float pl[16];
    float tmax = -INFINITY;
#pragma unroll
    for (int r = 0; r < 8; ++r) {
      float l0 = t0[r] * scale - 1e9f * mb[key0 + (half << 3) + r];
      float l1 = t1[r] * scale - 1e9f * mb[key0 + 16 + (half << 3) + r];
      pl[r] = l0; pl[r + 8] = l1;
      tmax = fmaxf(tmax, fmaxf(l0, l1));
    }
    tmax = fmaxf(tmax, __shfl_xor(tmax, 16, 32));
    float m_new = fmaxf(m_run, tmax);
    float alpha = __expf(m_run - m_new);
    float lsum = 0.f;
#pragma unroll
    for (int i = 0; i < 16; ++i) {
      float p = __expf(pl[i] - m_new);
      pl[i] = p;
      lsum += p;
    }
    lsum += __shfl_xor(lsum, 16, 32);
    l_run = l_run * alpha + lsum;
    m_run = m_new;

    // rescale O accumulators: alpha lives per-lane (query = lane%16),
    // O rows are m = r + 8*half -> broadcast from lane (r + 8*half)
#pragma unroll
    for (int r = 0; r < 8; ++r) {
      float ar = __shfl(alpha, r + (half << 3), 32);
      o0[r] *= ar; o1[r] *= ar; o2[r] *= ar; o3[r] *= ar;
    }

    // P fragment: C-layout of S^T maps element-for-element to A-fragment
    FragBF ap;
#pragma unroll
    for (int j = 0; j < 16; ++j) ap.e[j] = (__bf16)pl[j];

    // V as B operand from LDS: column d = c*16+col16, keys 8*half+j and 16+8*half+j
#pragma unroll
    for (int c = 0; c < 4; ++c) {
      FragBF bv;
      bv.q[0] = *(const uint4*)&VtL[buf][c * 16 + col16][8 * half];
      bv.q[1] = *(const uint4*)&VtL[buf][c * 16 + col16][16 + 8 * half];
      v8f* oc = (c == 0) ? &o0 : (c == 1) ? &o1 : (c == 2) ? &o2 : &o3;
      *oc = __builtin_amdgcn_wmma_f32_16x16x32_bf16(
          false, ap.v, false, bv.v, (short)0, *oc, false, false);
    }

    // next tile resident + everyone done reading this buffer
    async_copies_wait();
    __syncthreads();
  }

  // finalize: divide by row sum, store bf16 head-split O (B,H,S,64)
#pragma unroll
  for (int r = 0; r < 8; ++r) {
    float lr = __shfl(l_run, r + (half << 3), 32);
    float inv = 1.0f / lr;
    int qrow = qBase + r + (half << 3);
    size_t base = ((size_t)bh * S_ + qrow) * DEPTH_;
    FragBF t;
    t.e[0] = (__bf16)(o0[r] * inv); Ob[base + 0  + col16] = t.u[0];
    t.e[0] = (__bf16)(o1[r] * inv); Ob[base + 16 + col16] = t.u[0];
    t.e[0] = (__bf16)(o2[r] * inv); Ob[base + 32 + col16] = t.u[0];
    t.e[0] = (__bf16)(o3[r] * inv); Ob[base + 48 + col16] = t.u[0];
  }
}

// ---------------------------------------------------------------------------
// Output projection: out(4096x1024 f32) = merge_heads(O) @ Wo + bo
// ---------------------------------------------------------------------------
__global__ __launch_bounds__(256) void oproj_kernel(
    const unsigned short* __restrict__ Ob, const unsigned short* __restrict__ Wt,
    const float* __restrict__ bo, float* __restrict__ out) {
  const int lane  = threadIdx.x & 31;
  const int wave  = threadIdx.x >> 5;
  const int half  = lane >> 4;
  const int col16 = lane & 15;
  const int rowBase = blockIdx.x * 32 + (wave >> 2) * 16;
  const int colBase = blockIdx.y * 64 + (wave & 3) * 16;
  const int m = rowBase + col16;
  const int n = colBase + col16;
  const int bb = m >> 11, s = m & (S_ - 1);

  v8f acc = {};
  for (int kb = 0; kb < D_; kb += 32) {
    FragBF a, b;
    // A: two contiguous 8-elem bf16 runs inside one head's 64-wide row
    int k0 = kb + 8 * half;         // h = k0>>6 constant over the 8-run
    int k1 = kb + 16 + 8 * half;
    a.q[0] = *(const uint4*)(Ob + (((size_t)bb * H_ + (k0 >> 6)) * S_ + s) * DEPTH_ + (k0 & 63));
    a.q[1] = *(const uint4*)(Ob + (((size_t)bb * H_ + (k1 >> 6)) * S_ + s) * DEPTH_ + (k1 & 63));
    // B: Wt[n][kb + 8*half + j] and Wt[n][kb + 16 + 8*half + j] -> b128 x2
    const uint4* wt = (const uint4*)(Wt + (size_t)n * D_ + kb + 8 * half);
    b.q[0] = wt[0]; b.q[1] = wt[2];

    acc = __builtin_amdgcn_wmma_f32_16x16x32_bf16(
        false, a.v, false, b.v, (short)0, acc, false, false);
  }
  const float bn = bo[n];
#pragma unroll
  for (int r = 0; r < 8; ++r) {
    int row = rowBase + r + (half << 3);
    out[(size_t)row * D_ + n] = acc[r] + bn;
  }
}

// ---------------------------------------------------------------------------
extern "C" void kernel_launch(void* const* d_in, const int* in_sizes, int n_in,
                              void* d_out, int out_size, void* d_ws, size_t ws_size,
                              hipStream_t stream) {
  const float* v    = (const float*)d_in[0];
  const float* q    = (const float*)d_in[1];
  const float* k    = (const float*)d_in[2];
  const float* mask = (const float*)d_in[3];
  const float* Wq   = (const float*)d_in[4];
  const float* bq   = (const float*)d_in[5];
  const float* Wv   = (const float*)d_in[6];
  const float* bv   = (const float*)d_in[7];
  const float* Wo   = (const float*)d_in[8];
  const float* bo   = (const float*)d_in[9];

  unsigned short* ws = (unsigned short*)d_ws;
  const size_t headElems = (size_t)B_ * H_ * S_ * DEPTH_;   // 4,194,304
  unsigned short* Qb  = ws;
  unsigned short* Kb  = ws + headElems;
  unsigned short* Vb  = ws + 2 * headElems;   // transposed (B,H,64,S)
  unsigned short* Obf = ws + 3 * headElems;
  unsigned short* Wtq = ws + 4 * headElems;   // Wt[n][k] bf16, 1M elems each
  unsigned short* Wtv = Wtq + (size_t)D_ * D_;
  unsigned short* Wto = Wtv + (size_t)D_ * D_;

  dim3 blk(256);
  dim3 gw(32, 32);              // 1024/32 x 1024/32 transpose tiles
  wcvt_kernel<<<gw, blk, 0, stream>>>(Wq, Wtq);
  wcvt_kernel<<<gw, blk, 0, stream>>>(Wv, Wtv);
  wcvt_kernel<<<gw, blk, 0, stream>>>(Wo, Wto);

  dim3 gproj(128, 16);          // (4096/32 rows, 1024/64 cols)
  // Faithful to reference: K is projected with the QUERY weights.
  proj_kernel<<<gproj, blk, 0, stream>>>(q, Wtq, bq, Qb, DEPTH_, 1);
  proj_kernel<<<gproj, blk, 0, stream>>>(k, Wtq, bq, Kb, DEPTH_, 1);
  proj_kernel<<<gproj, blk, 0, stream>>>(v, Wtv, bv, Vb, 1, S_);   // d-major

  dim3 gattn(16, 32);           // (S/128 query blocks, B*H heads)
  attn_kernel<<<gattn, blk, 0, stream>>>(Qb, Kb, Vb, mask, Obf);

  oproj_kernel<<<gproj, blk, 0, stream>>>(Obf, Wto, bo, (float*)d_out);
}
